// NodeEdgeEarlyInteraction_42142219108479
// MI455X (gfx1250) — matrix operand order repaired and checked by reading, not compiled, not run on hardware
//
#include <hip/hip_runtime.h>
#include <hip/hip_bf16.h>
#include <cstring>
#include <cstdint>

typedef _Float16 half16 __attribute__((ext_vector_type(16)));
typedef float    float8 __attribute__((ext_vector_type(8)));

#define VN 6144      // 128 groups * 48 nodes
#define VE 24576     // 128 groups * 192 edges
// D=128, M=256, T=5, K=3, S=64, MAX_N=64, MAX_E=256, TEMP=0.1

// ---------------------------------------------------------------------------
// Generic WMMA GEMM:  C = epilogue( accScale * (A @ B) )
//   A: row-major, up to 2 column segments, each with optional row-index gather
//      (negative index -> reads the zero page, i.e. a zero row).  B: row-major
//      [K,N] with optional row index (negative -> zero row) staged through LDS
//      in fragment layout, or Bt: [N,K] row-major (computes A@Bt^T) loaded
//      directly (contiguous K).
//   Epilogue: + bias*biasScale + add*addScale, optional ReLU, optional
//   row-mask ((row % maskMod) >= maskLim -> 0), output store or indexed
//   scatter (negative -> skip) with optional f32 atomic add.
//   Batched via blockIdx.z with element strides.
//   Block: 256 threads = 8 wave32 (4 row-tiles x 2 col-groups).
//   CT = # of 16-col tiles per block (8 -> 64x128 tile, 4 -> 64x64 tile).
// ---------------------------------------------------------------------------
struct GemmDesc {
    const float* a0; long lda0; const int* aIdx0; long aBS0;
    const float* a1; long lda1; const int* aIdx1; long aBS1;
    long aIdxBS; int w0;
    const float* b;  long ldb;  long bBS; const int* bIdx; long bIdxBS;
    const float* bt; long ldbt; long btBS;
    float* c; long ldc; long cBS; const int* cIdx; long cIdxBS; int cAtomic;
    const float* bias; float biasScale;
    const float* add; long ldadd; long addBS; float addScale;
    float accScale;
    int relu; int maskMod; int maskLim;
    int M, N, K;
    const float* zrow;     // zero page (>= 2048 f32, zero-filled)
};

__device__ __forceinline__ void gemm_epilogue(const GemmDesc& g, int bz, int row0,
                                              int col, int hl, float8 acc) {
    float bv = 0.0f;
    if (g.bias) bv = g.bias[col] * g.biasScale;
    float* C = g.c + (size_t)bz * g.cBS;
    const float* ADD = g.add ? g.add + (size_t)bz * g.addBS : nullptr;
    const int* cIdx = g.cIdx ? g.cIdx + (size_t)bz * g.cIdxBS : nullptr;
#pragma unroll
    for (int i = 0; i < 8; ++i) {
        int rr = row0 + i + hl * 8;
        float v = acc[i] * g.accScale + bv;
        if (ADD) v += ADD[(size_t)rr * g.ldadd + col] * g.addScale;
        if (g.relu) v = fmaxf(v, 0.0f);
        if (g.maskMod && ((rr % g.maskMod) >= g.maskLim)) v = 0.0f;
        long orow = rr;
        if (cIdx) { orow = cIdx[rr]; if (orow < 0) continue; }
        float* dst = &C[(size_t)orow * g.ldc + col];
        if (g.cAtomic) unsafeAtomicAdd(dst, v);
        else *dst = v;
    }
}

// A fragment: 16x32 f16; lane half h holds K = h*16+j (j<16).
__device__ __forceinline__ half16 load_a_frag(const float* rowp0, const float* rowp1,
                                              int w0, int k0, int hl) {
    const float* rp; int kk;
    if (k0 < w0) { rp = rowp0; kk = k0; }
    else         { rp = rowp1; kk = k0 - w0; }
    const float* p = rp + kk + hl * 16;
    __builtin_prefetch(p + 32, 0, 3);   // next K-chunk of this row
    half16 af;
#pragma unroll
    for (int j = 0; j < 8; ++j) af[j]     = (_Float16)p[j];
#pragma unroll
    for (int j = 0; j < 8; ++j) af[8 + j] = (_Float16)p[8 + j];
    return af;
}

template <int CT>
__global__ __launch_bounds__(256) void wmma_gemm_kernel(GemmDesc g) {
    constexpr int NT = CT / 2;                 // 16-col tiles per wave
    __shared__ _Float16 bsm[CT * 512];         // [ct][h][n][j] halves (32B frags)

    const int tid  = threadIdx.x;
    const int lane = tid & 31;
    const int wave = tid >> 5;
    const int hl   = lane >> 4;                // half-wave (0 or 1)
    const int lm   = lane & 15;
    const int row0 = blockIdx.y * 64 + (wave >> 1) * 16;
    const int wc   = wave & 1;                 // col wave-group
    const int colBlk = blockIdx.x * (CT * 16);
    const int bz   = blockIdx.z;

    // Per-lane A row pointers (fragment row = row0 + lm), hoisted from K loop.
    // Negative gather index -> zero page, so the hot loop is branch-free.
    const int r = row0 + lm;
    const float* rowp0;
    const float* rowp1 = g.zrow;
    {
        const float* A0 = g.a0 + (size_t)bz * g.aBS0;
        int ri = r;
        if (g.aIdx0) ri = g.aIdx0[(size_t)bz * g.aIdxBS + r];
        rowp0 = (ri >= 0) ? A0 + (size_t)ri * g.lda0 : g.zrow;
    }
    if (g.a1) {
        const float* A1 = g.a1 + (size_t)bz * g.aBS1;
        int ri = r;
        if (g.aIdx1) ri = g.aIdx1[(size_t)bz * g.aIdxBS + r];
        rowp1 = (ri >= 0) ? A1 + (size_t)ri * g.lda1 : g.zrow;
    }

    float8 acc[NT];
    {
        float8 z = {};
#pragma unroll
        for (int i = 0; i < NT; ++i) acc[i] = z;
    }

    if (g.bt) {
        // ---- A @ Bt^T path: direct contiguous loads, no LDS, no barriers.
        const float* BT = g.bt + (size_t)bz * g.btBS;
        const float* btp[NT];
#pragma unroll
        for (int i = 0; i < NT; ++i)
            btp[i] = BT + (size_t)(colBlk + (wc * NT + i) * 16 + lm) * g.ldbt;
        for (int k0 = 0; k0 < g.K; k0 += 32) {
            half16 af = load_a_frag(rowp0, rowp1, g.w0, k0, hl);
#pragma unroll
            for (int i = 0; i < NT; ++i) {
                const float* p = btp[i] + k0 + hl * 16;   // lane half h: K = h*16+j
                half16 bf;
#pragma unroll
                for (int j = 0; j < 16; ++j) bf[j] = (_Float16)p[j];
                acc[i] = __builtin_amdgcn_wmma_f32_16x16x32_f16(
                    false, af, false, bf, (short)0, acc[i], false, false);
            }
        }
    } else {
        // ---- Row-major B path: cooperative LDS staging in fragment layout.
        const float* Bp   = g.b + (size_t)bz * g.bBS;
        const int*   bIdx = g.bIdx ? g.bIdx + (size_t)bz * g.bIdxBS : nullptr;
        for (int k0 = 0; k0 < g.K; k0 += 32) {
            half16 af = load_a_frag(rowp0, rowp1, g.w0, k0, hl);
            __syncthreads();
            // bsm[ct*512 + h*256 + n*16 + j] = B[k0+h*16+j][colBlk+ct*16+n]
            for (int t = tid; t < 32 * CT; t += 256) {
                int kk = t / CT;              // 0..31 within chunk
                int cs = t % CT;              // 16-col tile
                int krow = k0 + kk;
                int ks = bIdx ? bIdx[krow] : krow;
                const float* rb = (ks >= 0)
                    ? Bp + (size_t)ks * g.ldb + colBlk + cs * 16 : g.zrow;
                int h = kk >> 4, j = kk & 15;
                _Float16* dst = &bsm[cs * 512 + h * 256 + j];
#pragma unroll
                for (int n = 0; n < 16; ++n) dst[n * 16] = (_Float16)rb[n];
            }
            __syncthreads();
#pragma unroll
            for (int i = 0; i < NT; ++i) {
                int ct = wc * NT + i;
                half16 bf = *reinterpret_cast<const half16*>(
                    &bsm[ct * 512 + hl * 256 + lm * 16]);
                acc[i] = __builtin_amdgcn_wmma_f32_16x16x32_f16(
                    false, af, false, bf, (short)0, acc[i], false, false);
            }
        }
    }
#pragma unroll
    for (int i = 0; i < NT; ++i)
        gemm_epilogue(g, bz, row0, colBlk + (wc * NT + i) * 16 + lm, hl, acc[i]);
}

// ---------------------------------------------------------------------------
// Small helper kernels
// ---------------------------------------------------------------------------
__global__ void zero_kernel(float* p, size_t n) {
    size_t t = (size_t)blockIdx.x * 256 + threadIdx.x;
    if (t < n) p[t] = 0.0f;
}

__global__ void exp_kernel(float* p, size_t n) {
    size_t t = (size_t)blockIdx.x * 256 + threadIdx.x;
    if (t < n) p[t] = expf(p[t]);
}

// padN[g*64+r]  = g*48 +r if r<48  else -1   (8192 entries)
// padE[g*256+r] = g*192+r if r<192 else -1   (32768 entries)
__global__ void build_pad_kernel(int* padN, int* padE) {
    int t = blockIdx.x * 256 + threadIdx.x;
    if (t < 8192)  { int gg = t >> 6, rr = t & 63;  padN[t] = (rr < 48)  ? gg * 48  + rr : -1; }
    if (t < 32768) { int gg = t >> 8, rr = t & 255; padE[t] = (rr < 192) ? gg * 192 + rr : -1; }
}

// Wab = msg_W[0:128,:] + msg_W[128:256,:]
__global__ void wab_kernel(float* wab, const float* msgW) {
    int t = blockIdx.x * 256 + threadIdx.x;
    if (t < 128 * 256) wab[t] = msgW[t] + msgW[128 * 256 + t];
}

// hsum[e,:] = h[from[e],:] + h[to[e],:]   (128 cols, h has leading dim ldh)
__global__ void hsum_kernel(float* hs, const float* h, long ldh,
                            const int* fi, const int* ti) {
    size_t t = (size_t)blockIdx.x * 256 + threadIdx.x;
    if (t >= (size_t)VE * 128) return;
    int e = (int)(t >> 7), d = (int)(t & 127);
    hs[t] = h[(size_t)fi[e] * ldh + d] + h[(size_t)ti[e] * ldh + d];
}

// In-place Sinkhorn: 10 iters of (row-LSE subtract, col-LSE subtract),
// one workgroup per group; matrix stays in global (L2-resident).
template <int MS>
__global__ __launch_bounds__(256) void sinkhorn_kernel(float* la) {
    float* p = la + (size_t)blockIdx.x * MS * MS;
    for (int it = 0; it < 10; ++it) {
        for (int q = threadIdx.x; q < MS; q += 256) {
            float* row = p + (size_t)q * MS;
            float m = -1e30f;
            for (int c = 0; c < MS; ++c) m = fmaxf(m, row[c]);
            float s = 0.0f;
            for (int c = 0; c < MS; ++c) s += expf(row[c] - m);
            float lse = m + logf(s);
            for (int c = 0; c < MS; ++c) row[c] -= lse;
        }
        __syncthreads();
        for (int c = threadIdx.x; c < MS; c += 256) {
            float m = -1e30f;
            for (int q = 0; q < MS; ++q) m = fmaxf(m, p[(size_t)q * MS + c]);
            float s = 0.0f;
            for (int q = 0; q < MS; ++q) s += expf(p[(size_t)q * MS + c] - m);
            float lse = m + logf(s);
            for (int q = 0; q < MS; ++q) p[(size_t)q * MS + c] -= lse;
        }
        __syncthreads();
    }
}

// Batched square transpose (batch = blockIdx.y)
__global__ void transpose_kernel(float* dst, const float* src, int n) {
    const float* S = src + (size_t)blockIdx.y * n * n;
    float*       D = dst + (size_t)blockIdx.y * n * n;
    int t = blockIdx.x * 256 + threadIdx.x;
    if (t < n * n) {
        int rr = t / n, cc = t % n;
        D[(size_t)cc * n + rr] = S[t];
    }
}

// score[b] = -sum_{q<64,d<128} relu(ffq[q,d] - sum_{c<48} plan[q,c]*ffc[c,d])
__global__ __launch_bounds__(256) void score_kernel(const float* plan,
                                                    const float* updn, float* out) {
    int b = blockIdx.x;
    const float* P = plan + (size_t)b * 64 * 64;
    __shared__ float red[256];
    float local = 0.0f;
    for (int t = threadIdx.x; t < 64 * 128; t += 256) {
        int q = t >> 7, d = t & 127;
        float qi = 0.0f;
        for (int c = 0; c < 48; ++c)
            qi += P[q * 64 + c] * updn[(size_t)((2 * b + 1) * 48 + c) * 768 + 640 + d];
        float fq = (q < 48) ? updn[(size_t)(2 * b * 48 + q) * 768 + 640 + d] : 0.0f;
        local += fmaxf(fq - qi, 0.0f);
    }
    red[threadIdx.x] = local; __syncthreads();
    for (int s = 128; s > 0; s >>= 1) {
        if (threadIdx.x < s) red[threadIdx.x] += red[threadIdx.x + s];
        __syncthreads();
    }
    if (threadIdx.x == 0) out[b] = -red[0];
}

// ---------------------------------------------------------------------------
// Host orchestration
// ---------------------------------------------------------------------------
static const float* g_zpage = nullptr;

static GemmDesc gd_init() {
    GemmDesc g;
    memset(&g, 0, sizeof(g));
    g.biasScale = 1.0f; g.addScale = 1.0f; g.accScale = 1.0f;
    g.zrow = g_zpage;
    return g;
}
static void G(hipStream_t s, GemmDesc& g, int batch = 1) {
    if (g.w0 == 0) g.w0 = g.K;
    if ((g.N & 127) == 0) {
        dim3 grid((unsigned)(g.N / 128), (unsigned)(g.M / 64), (unsigned)batch);
        wmma_gemm_kernel<8><<<grid, dim3(256), 0, s>>>(g);
    } else {
        dim3 grid((unsigned)(g.N / 64), (unsigned)(g.M / 64), (unsigned)batch);
        wmma_gemm_kernel<4><<<grid, dim3(256), 0, s>>>(g);
    }
}

extern "C" void kernel_launch(void* const* d_in, const int* in_sizes, int n_in,
                              void* d_out, int out_size, void* d_ws, size_t ws_size,
                              hipStream_t stream) {
    (void)in_sizes; (void)n_in; (void)out_size; (void)ws_size;
    const float* node_features = (const float*)d_in[0];
    const float* edge_features = (const float*)d_in[1];
    const int*   from_idx = (const int*)d_in[2];
    const int*   to_idx   = (const int*)d_in[3];
    const float* enc_node_W = (const float*)d_in[4];
    const float* enc_node_b = (const float*)d_in[5];
    const float* enc_edge_W = (const float*)d_in[6];
    const float* enc_edge_b = (const float*)d_in[7];
    const float* ni_W1 = (const float*)d_in[8];
    const float* ni_b1 = (const float*)d_in[9];
    const float* ni_W2 = (const float*)d_in[10];
    const float* ni_b2 = (const float*)d_in[11];
    const float* ei_W1 = (const float*)d_in[12];
    const float* ei_b1 = (const float*)d_in[13];
    const float* ei_W2 = (const float*)d_in[14];
    const float* ei_b2 = (const float*)d_in[15];
    const float* msg_W = (const float*)d_in[16];
    const float* msg_b = (const float*)d_in[17];
    const float* nu_W1 = (const float*)d_in[18];
    const float* nu_b1 = (const float*)d_in[19];
    const float* nu_W2 = (const float*)d_in[20];
    const float* nu_b2 = (const float*)d_in[21];
    const float* ns_W1 = (const float*)d_in[22];
    const float* ns_b1 = (const float*)d_in[23];
    const float* ns_W2 = (const float*)d_in[24];
    const float* ns_b2 = (const float*)d_in[25];
    const float* es_W1 = (const float*)d_in[26];
    const float* es_b1 = (const float*)d_in[27];
    const float* es_W2 = (const float*)d_in[28];
    const float* es_b2 = (const float*)d_in[29];

    float* Wf = (float*)d_ws;
    size_t off = 0;
    auto alloc = [&](size_t n) { float* p = Wf + off; off += n; return p; };
    float* zpage      = alloc((size_t)2048);
    float* enc_n      = alloc((size_t)VN * 128);
    float* enc_e      = alloc((size_t)VE * 128);
    float* node_store = alloc((size_t)VN * 768);
    float* edge_store = alloc((size_t)VE * 1536);
    float* upd_node   = alloc((size_t)VN * 768);
    float* upd_edge   = alloc((size_t)VE * 1536);
    float* tmpN       = alloc((size_t)VN * 256);
    float* tmpE       = alloc((size_t)VE * 384);
    float* h_comb     = alloc((size_t)VN * 128);
    float* e_comb     = alloc((size_t)VE * 128);
    float* Ecb        = alloc((size_t)VE * 256);
    float* agg        = alloc((size_t)VN * 256);
    float* hsum       = alloc((size_t)VE * 128);
    float* Wab        = alloc((size_t)128 * 256);
    float* tTmpN      = alloc((size_t)8192 * 64);
    float* tN         = alloc((size_t)8192 * 64);
    float* tTmpE      = alloc((size_t)32768 * 64);
    float* tE         = alloc((size_t)32768 * 64);
    float* laN        = alloc((size_t)64 * 64 * 64);
    float* planTN     = alloc((size_t)64 * 64 * 64);
    float* laE        = alloc((size_t)64 * 256 * 256);
    float* planTE     = alloc((size_t)64 * 256 * 256);
    int*   padN       = (int*)alloc(8192);
    int*   padE       = (int*)alloc(32768);

    g_zpage = zpage;

    auto zero = [&](float* p, size_t n) {
        zero_kernel<<<dim3((unsigned)((n + 255) / 256)), dim3(256), 0, stream>>>(p, n);
    };

    zero(zpage, 2048);
    build_pad_kernel<<<dim3(128), dim3(256), 0, stream>>>(padN, padE);
    wab_kernel<<<dim3(128), dim3(256), 0, stream>>>(Wab, msg_W);
    zero(node_store, (size_t)VN * 768);
    zero(edge_store, (size_t)VE * 1536);
    zero(upd_node,   (size_t)VN * 768);   // slot 0 must stay zero
    zero(upd_edge,   (size_t)VE * 1536);

    // Encoders
    { GemmDesc g = gd_init(); g.a0 = node_features; g.lda0 = 32; g.b = enc_node_W; g.ldb = 128;
      g.bias = enc_node_b; g.c = enc_n; g.ldc = 128; g.M = VN; g.N = 128; g.K = 32; G(stream, g); }
    { GemmDesc g = gd_init(); g.a0 = edge_features; g.lda0 = 32; g.b = enc_edge_W; g.ldb = 128;
      g.bias = enc_edge_b; g.c = enc_e; g.ldc = 128; g.M = VE; g.N = 128; g.K = 32; G(stream, g); }

    for (int kk = 0; kk < 3; ++kk) {
        for (int p = 1; p <= 5; ++p) {
            const float* hPtr; long hLd;
            if (p == 1) { hPtr = enc_n; hLd = 128; }
            else        { hPtr = upd_node + (size_t)(p - 1) * 128; hLd = 768; }
            zero(agg, (size_t)VN * 256);
            // h_comb = MLP2([h, node_store slot p-1])
            { GemmDesc g = gd_init(); g.a0 = hPtr; g.lda0 = hLd; g.w0 = 128;
              g.a1 = node_store + (size_t)(p - 1) * 128; g.lda1 = 768;
              g.b = ni_W1; g.ldb = 256; g.bias = ni_b1; g.relu = 1;
              g.c = tmpN; g.ldc = 256; g.M = VN; g.N = 256; g.K = 256; G(stream, g); }
            { GemmDesc g = gd_init(); g.a0 = tmpN; g.lda0 = 256;
              g.b = ni_W2; g.ldb = 128; g.bias = ni_b2;
              g.c = h_comb; g.ldc = 128; g.M = VN; g.N = 128; g.K = 256; G(stream, g); }
            // e_comb = MLP2([enc_e, edge_store slot p-1])
            { GemmDesc g = gd_init(); g.a0 = enc_e; g.lda0 = 128; g.w0 = 128;
              g.a1 = edge_store + (size_t)(p - 1) * 256; g.lda1 = 1536;
              g.b = ei_W1; g.ldb = 384; g.bias = ei_b1; g.relu = 1;
              g.c = tmpE; g.ldc = 384; g.M = VE; g.N = 384; g.K = 384; G(stream, g); }
            { GemmDesc g = gd_init(); g.a0 = tmpE; g.lda0 = 384;
              g.b = ei_W2; g.ldb = 128; g.bias = ei_b2;
              g.c = e_comb; g.ldc = 128; g.M = VE; g.N = 128; g.K = 384; G(stream, g); }
            // Ec = e_comb @ Wc (rows 256..383 of msg_W)
            { GemmDesc g = gd_init(); g.a0 = e_comb; g.lda0 = 128;
              g.b = msg_W + (size_t)256 * 256; g.ldb = 256;
              g.c = Ecb; g.ldc = 256; g.M = VE; g.N = 256; g.K = 128; G(stream, g); }
            // fwd: [src,dst]@[Wa;Wb] + b + Ec  -> atomic scatter into agg[to]
            { GemmDesc g = gd_init(); g.a0 = h_comb; g.lda0 = 128; g.aIdx0 = from_idx;
              g.a1 = h_comb; g.lda1 = 128; g.aIdx1 = to_idx; g.w0 = 128;
              g.b = msg_W; g.ldb = 256; g.bias = msg_b;
              g.add = Ecb; g.ldadd = 256;
              g.c = agg; g.ldc = 256; g.cIdx = to_idx; g.cAtomic = 1;
              g.M = VE; g.N = 256; g.K = 256; G(stream, g); }
            // bwd: [dst,src]@[Wa;Wb] + b + Ec  -> atomic scatter into agg[from]
            { GemmDesc g = gd_init(); g.a0 = h_comb; g.lda0 = 128; g.aIdx0 = to_idx;
              g.a1 = h_comb; g.lda1 = 128; g.aIdx1 = from_idx; g.w0 = 128;
              g.b = msg_W; g.ldb = 256; g.bias = msg_b;
              g.add = Ecb; g.ldadd = 256;
              g.c = agg; g.ldc = 256; g.cIdx = from_idx; g.cAtomic = 1;
              g.M = VE; g.N = 256; g.K = 256; G(stream, g); }
            // h = MLP2([h_comb, agg]) -> upd_node slot p
            { GemmDesc g = gd_init(); g.a0 = h_comb; g.lda0 = 128; g.w0 = 128;
              g.a1 = agg; g.lda1 = 256;
              g.b = nu_W1; g.ldb = 256; g.bias = nu_b1; g.relu = 1;
              g.c = tmpN; g.ldc = 256; g.M = VN; g.N = 256; g.K = 384; G(stream, g); }
            { GemmDesc g = gd_init(); g.a0 = tmpN; g.lda0 = 256;
              g.b = nu_W2; g.ldb = 128; g.bias = nu_b2;
              g.c = upd_node + (size_t)p * 128; g.ldc = 768;
              g.M = VN; g.N = 128; g.K = 256; G(stream, g); }
            // msgs = (src2+dst2)@(Wa+Wb) + 2b + 2Ec -> upd_edge slot p
            hsum_kernel<<<dim3((unsigned)(((size_t)VE * 128 + 255) / 256)), dim3(256), 0, stream>>>(
                hsum, upd_node + (size_t)p * 128, 768, from_idx, to_idx);
            { GemmDesc g = gd_init(); g.a0 = hsum; g.lda0 = 128;
              g.b = Wab; g.ldb = 256; g.bias = msg_b; g.biasScale = 2.0f;
              g.add = Ecb; g.ldadd = 256; g.addScale = 2.0f;
              g.c = upd_edge + (size_t)p * 256; g.ldc = 1536;
              g.M = VE; g.N = 256; g.K = 128; G(stream, g); }
        }

        // ---------------- node transport ----------------
        { GemmDesc g = gd_init(); g.a0 = upd_node + 640; g.lda0 = 768; g.aIdx0 = padN;
          g.b = ns_W1; g.ldb = 64; g.bias = ns_b1; g.relu = 1;
          g.c = tTmpN; g.ldc = 64; g.M = 8192; g.N = 64; g.K = 128; G(stream, g); }
        { GemmDesc g = gd_init(); g.a0 = tTmpN; g.lda0 = 64;
          g.b = ns_W2; g.ldb = 64; g.bias = ns_b2; g.maskMod = 64; g.maskLim = 48;
          g.c = tN; g.ldc = 64; g.M = 8192; g.N = 64; g.K = 64; G(stream, g); }
        { GemmDesc g = gd_init(); g.a0 = tN; g.lda0 = 64; g.aBS0 = 8192;
          g.bt = tN + 4096; g.ldbt = 64; g.btBS = 8192; g.accScale = 10.0f; // 1/TEMP
          g.c = laN; g.ldc = 64; g.cBS = 4096; g.M = 64; g.N = 64; g.K = 64; G(stream, g, 64); }
        sinkhorn_kernel<64><<<dim3(64), dim3(256), 0, stream>>>(laN);
        exp_kernel<<<dim3((262144 + 255) / 256), dim3(256), 0, stream>>>(laN, 262144);
        transpose_kernel<<<dim3(16, 64), dim3(256), 0, stream>>>(planTN, laN, 64);
        // qi = plan @ cand_features -> node_store (query rows)
        { GemmDesc g = gd_init(); g.a0 = laN; g.lda0 = 64; g.aBS0 = 4096;
          g.b = upd_node + 128; g.ldb = 768; g.bIdx = padN + 64; g.bIdxBS = 128;
          g.c = node_store + 128; g.ldc = 768; g.cIdx = padN; g.cIdxBS = 128;
          g.M = 64; g.N = 640; g.K = 64; G(stream, g, 64); }
        // ci = plan^T @ query_features -> node_store (candidate rows)
        { GemmDesc g = gd_init(); g.a0 = planTN; g.lda0 = 64; g.aBS0 = 4096;
          g.b = upd_node + 128; g.ldb = 768; g.bIdx = padN; g.bIdxBS = 128;
          g.c = node_store + 128; g.ldc = 768; g.cIdx = padN + 64; g.cIdxBS = 128;
          g.M = 64; g.N = 640; g.K = 64; G(stream, g, 64); }

        // ---------------- edge transport ----------------
        { GemmDesc g = gd_init(); g.a0 = upd_edge + 1280; g.lda0 = 1536; g.aIdx0 = padE;
          g.b = es_W1; g.ldb = 64; g.bias = es_b1; g.relu = 1;
          g.c = tTmpE; g.ldc = 64; g.M = 32768; g.N = 64; g.K = 256; G(stream, g); }
        { GemmDesc g = gd_init(); g.a0 = tTmpE; g.lda0 = 64;
          g.b = es_W2; g.ldb = 64; g.bias = es_b2; g.maskMod = 256; g.maskLim = 192;
          g.c = tE; g.ldc = 64; g.M = 32768; g.N = 64; g.K = 64; G(stream, g); }
        { GemmDesc g = gd_init(); g.a0 = tE; g.lda0 = 64; g.aBS0 = 32768;
          g.bt = tE + 16384; g.ldbt = 64; g.btBS = 32768; g.accScale = 10.0f;
          g.c = laE; g.ldc = 256; g.cBS = 65536; g.M = 256; g.N = 256; g.K = 64; G(stream, g, 64); }
        sinkhorn_kernel<256><<<dim3(64), dim3(256), 0, stream>>>(laE);
        exp_kernel<<<dim3((unsigned)(((size_t)4194304 + 255) / 256)), dim3(256), 0, stream>>>(
            laE, (size_t)4194304);
        transpose_kernel<<<dim3(256, 64), dim3(256), 0, stream>>>(planTE, laE, 256);
        { GemmDesc g = gd_init(); g.a0 = laE; g.lda0 = 256; g.aBS0 = 65536;
          g.b = upd_edge + 256; g.ldb = 1536; g.bIdx = padE + 256; g.bIdxBS = 512;
          g.c = edge_store + 256; g.ldc = 1536; g.cIdx = padE; g.cIdxBS = 512;
          g.M = 256; g.N = 1280; g.K = 256; G(stream, g, 64); }
        { GemmDesc g = gd_init(); g.a0 = planTE; g.lda0 = 256; g.aBS0 = 65536;
          g.b = upd_edge + 256; g.ldb = 1536; g.bIdx = padE; g.bIdxBS = 512;
          g.c = edge_store + 256; g.ldc = 1536; g.cIdx = padE + 256; g.cIdxBS = 512;
          g.M = 256; g.N = 1280; g.K = 256; G(stream, g, 64); }
    }

    score_kernel<<<dim3(64), dim3(256), 0, stream>>>(laN, upd_node, (float*)d_out);
}
